// CMAF_23673859736343
// MI455X (gfx1250) — matrix operand order repaired and checked by reading, not compiled, not run on hardware
//
#include <hip/hip_runtime.h>

// CDNA5 / gfx1250 fused kernel for the 3-stream modality-attention block.
// Strategy: f16 WMMA (16x16x32, f32 accum) for every GEMM; weights pre-swizzled
// once into B-fragment layout in d_ws (~1.2 MB, L2-resident); one wave32 per
// 16-row tile does the whole chain fused in LDS (projections -> LN -> 2-key
// attention -> out-proj -> LN -> FFN+GELU -> LN -> softmax gating).

typedef __attribute__((ext_vector_type(16))) _Float16 v16h;
typedef __attribute__((ext_vector_type(8)))  float    v8f;

#define DIM      128
#define FFN_DIM  256
#define SPATIAL_ 1280

struct Smem {
  _Float16 P[3][16][128];   // post-LN projections (f16)   12 KB
  _Float16 q [16][128];     // q, later attention o, later x1
  _Float16 k0[16][128];     // k(kv0); with k1 reused as FFN hidden h[16][256]
  _Float16 k1[16][128];
  _Float16 v0[16][128];
  _Float16 v1[16][128];
  _Float16 x2[3][16][128];  // post-FFN LN per stream      12 KB
  float    stage[16][128];  // f32 row-major staging for LayerNorm
  _Float16 chunk[16][32];   // f16 staging of global activation K-chunk
  float    red [32][4];     // reduction scratch
  float    misc[16][4];     // per-row mu/inv or gates
};

union HFrag { v16h h; unsigned int u[8]; };

__device__ __forceinline__ v8f wmma_f16(v16h a, v16h b, v8f c) {
  return __builtin_amdgcn_wmma_f32_16x16x32_f16(false, a, false, b, (short)0, c,
                                                false, false);
}

// Gather a 16x32 f16 A-fragment from row-major LDS (ld halves per row).
// ISA 16-bit A layout: lane L (M = L&15), VGPR v holds K-pair
//   k = (v>>2)*16 + (L>>4)*8 + (v&3)*2  (+0/+1)
__device__ __forceinline__ v16h ldsA(const _Float16* buf, int ld, int kofs, int lane) {
  const int m = lane & 15, hi = lane >> 4;
  HFrag f;
#pragma unroll
  for (int v = 0; v < 8; ++v) {
    int k = ((v >> 2) << 4) + (hi << 3) + ((v & 3) << 1);
    f.u[v] = *(const unsigned int*)(buf + m * ld + kofs + k);
  }
  return f.h;
}

__device__ __forceinline__ void zero8(v8f* acc) {
  v8f z = {};
#pragma unroll
  for (int j = 0; j < 8; ++j) acc[j] = z;
}

// D-fragment layout: lane L holds column n = L&15 of rows M = (L>>4)*8 + v.
__device__ __forceinline__ void addbias(v8f* acc, const float* bias, int col0, int lane) {
  const int n = lane & 15;
#pragma unroll
  for (int j = 0; j < 8; ++j) {
    float bb = bias[col0 + j * 16 + n];
#pragma unroll
    for (int v = 0; v < 8; ++v) acc[j][v] += bb;
  }
}

__device__ __forceinline__ void gelu8(v8f* acc) {
#pragma unroll
  for (int j = 0; j < 8; ++j)
#pragma unroll
    for (int v = 0; v < 8; ++v) {
      float x = acc[j][v];
      acc[j][v] = 0.5f * x * (1.f + erff(x * 0.70710678118654752f));
    }
}

__device__ __forceinline__ void addres_h(v8f* acc, const _Float16* buf, int lane) {
  const int n = lane & 15, hi = lane >> 4;
#pragma unroll
  for (int j = 0; j < 8; ++j)
#pragma unroll
    for (int v = 0; v < 8; ++v)
      acc[j][v] += (float)buf[(hi * 8 + v) * 128 + j * 16 + n];
}

__device__ __forceinline__ void stage_f32(const v8f* acc, Smem* sm, int lane) {
  const int n = lane & 15, hi = lane >> 4;
#pragma unroll
  for (int j = 0; j < 8; ++j)
#pragma unroll
    for (int v = 0; v < 8; ++v)
      sm->stage[hi * 8 + v][j * 16 + n] = acc[j][v];
  __syncthreads();
}

__device__ __forceinline__ void store_h(const v8f* acc, _Float16* buf, int ld,
                                        int col0, int lane) {
  const int n = lane & 15, hi = lane >> 4;
#pragma unroll
  for (int j = 0; j < 8; ++j)
#pragma unroll
    for (int v = 0; v < 8; ++v)
      buf[(hi * 8 + v) * ld + col0 + j * 16 + n] = (_Float16)acc[j][v];
}

// LayerNorm of sm->stage (16x128 f32) -> f16 out (16x128), + optional emb add.
__device__ __forceinline__ void ln16(Smem* sm, const float* g, const float* b,
                                     const float* emb, _Float16* out, int lane) {
  const int r = lane & 15, hp = lane >> 4, c0 = hp * 64;
  const float* row = &sm->stage[r][0];
  float s = 0.f, s2 = 0.f;
  for (int c = c0; c < c0 + 64; ++c) { float x = row[c]; s += x; s2 += x * x; }
  sm->red[lane][0] = s; sm->red[lane][1] = s2;
  __syncthreads();
  if (lane < 16) {
    float ts = sm->red[lane][0] + sm->red[lane + 16][0];
    float t2 = sm->red[lane][1] + sm->red[lane + 16][1];
    float mu  = ts * (1.f / 128.f);
    float var = t2 * (1.f / 128.f) - mu * mu;
    sm->misc[lane][0] = mu;
    sm->misc[lane][1] = rsqrtf(var + 1e-5f);
  }
  __syncthreads();
  float mu = sm->misc[r][0], inv = sm->misc[r][1];
  for (int c = c0; c < c0 + 64; ++c) {
    float x = (row[c] - mu) * inv * g[c] + b[c];
    if (emb) x += emb[c];
    out[r * 128 + c] = (_Float16)x;
  }
  __syncthreads();
}

// GEMM with A from LDS f16 (16 x Kblk*32, stride ld), B fragments from global ws.
__device__ __forceinline__ void gemm_lds(const _Float16* act, int ld, int Kblk,
                                         const v16h* __restrict__ w, int njAll,
                                         int j0, v8f* acc, int lane) {
  for (int kb = 0; kb < Kblk; ++kb) {
    v16h a = ldsA(act, ld, kb * 32, lane);
    if (kb + 1 < Kblk)
      __builtin_prefetch(&w[((kb + 1) * njAll + j0) * 32 + lane], 0, 1);
#pragma unroll
    for (int j = 0; j < 8; ++j)
      acc[j] = wmma_f16(a, w[(kb * njAll + j0 + j) * 32 + lane], acc[j]);
  }
}

// GEMM with A streamed from global fp32 (convert->LDS chunk), 8 col-tiles.
__device__ __forceinline__ void gemm_glb(const float* __restrict__ act, long row0,
                                         int ldact, int Kblk,
                                         const v16h* __restrict__ w, v8f* acc,
                                         Smem* sm, int lane) {
  const int r = lane >> 1, cb = (lane & 1) * 16;
  for (int kb = 0; kb < Kblk; ++kb) {
    const float4* s4 = (const float4*)(act + (row0 + r) * (long)ldact + kb * 32 + cb);
    float4 f0 = s4[0], f1 = s4[1], f2 = s4[2], f3 = s4[3];
    _Float16* c = &sm->chunk[r][cb];
    c[0]=(_Float16)f0.x;  c[1]=(_Float16)f0.y;  c[2]=(_Float16)f0.z;  c[3]=(_Float16)f0.w;
    c[4]=(_Float16)f1.x;  c[5]=(_Float16)f1.y;  c[6]=(_Float16)f1.z;  c[7]=(_Float16)f1.w;
    c[8]=(_Float16)f2.x;  c[9]=(_Float16)f2.y;  c[10]=(_Float16)f2.z; c[11]=(_Float16)f2.w;
    c[12]=(_Float16)f3.x; c[13]=(_Float16)f3.y; c[14]=(_Float16)f3.z; c[15]=(_Float16)f3.w;
    __syncthreads();
    v16h a = ldsA(&sm->chunk[0][0], 32, 0, lane);
    if (kb + 1 < Kblk)
      __builtin_prefetch(&w[((kb + 1) * 8) * 32 + lane], 0, 1);
#pragma unroll
    for (int j = 0; j < 8; ++j)
      acc[j] = wmma_f16(a, w[(kb * 8 + j) * 32 + lane], acc[j]);
    __syncthreads();
  }
}

// Pre-swizzle a (Nout x K) fp32 row-major weight into f16 B-fragments:
// fragment (kb, j) holds W^T[kb*32 .. +32, j*16 .. +16]; lane L stores 16
// halves = K (L>>4)*8 + {0..7, 16..23} of column j*16+(L&15)  (32 B / lane).
__global__ __launch_bounds__(256) void swizzle_w(const float* __restrict__ src,
                                                 v16h* __restrict__ dst,
                                                 int K, int NJ, int KB) {
  int gid = blockIdx.x * blockDim.x + threadIdx.x;
  int total = KB * NJ * 32;
  if (gid >= total) return;
  int lane = gid & 31;
  int frag = gid >> 5;
  int j  = frag % NJ;
  int kb = frag / NJ;
  int ncol  = j * 16 + (lane & 15);
  int kbase = kb * 32 + ((lane >> 4) << 3);
  HFrag f;
#pragma unroll
  for (int i = 0; i < 16; ++i)
    f.h[i] = (_Float16)src[(long)ncol * K + kbase + (i & 7) + ((i >> 3) << 4)];
  dst[gid] = f.h;
}

__global__ __launch_bounds__(32) void cmaf_fused(
    const float* __restrict__ xs, const float* __restrict__ xg, const float* __restrict__ xf,
    const float* __restrict__ proj_b, const float* __restrict__ pln_g,
    const float* __restrict__ pln_b, const float* __restrict__ mod_emb,
    const float* __restrict__ in_b, const float* __restrict__ out_b,
    const float* __restrict__ aln_g, const float* __restrict__ aln_b,
    const float* __restrict__ ffn_b1, const float* __restrict__ ffn_b2,
    const float* __restrict__ fln_g, const float* __restrict__ fln_b,
    const float* __restrict__ gate_w, const float* __restrict__ gate_b,
    const v16h* __restrict__ wsp, const v16h* __restrict__ wgf,
    const v16h* __restrict__ wq,  const v16h* __restrict__ wk,
    const v16h* __restrict__ wv,  const v16h* __restrict__ wo,
    const v16h* __restrict__ wf1, const v16h* __restrict__ wf2,
    float* __restrict__ out) {
  __shared__ Smem sm;
  const int lane = threadIdx.x;
  const long row0 = (long)blockIdx.x * 16;
  v8f acc[8];

  // ---------------- projections -> P[0..2] ----------------
  zero8(acc);
  gemm_glb(xs, row0, SPATIAL_, 40, wsp, acc, &sm, lane);
  addbias(acc, proj_b + 0 * DIM, 0, lane);
  stage_f32(acc, &sm, lane);
  ln16(&sm, pln_g + 0 * DIM, pln_b + 0 * DIM, mod_emb + 0 * DIM, &sm.P[0][0][0], lane);

  zero8(acc);
  gemm_glb(xg, row0, DIM, 4, wgf, acc, &sm, lane);
  addbias(acc, proj_b + 1 * DIM, 0, lane);
  stage_f32(acc, &sm, lane);
  ln16(&sm, pln_g + 1 * DIM, pln_b + 1 * DIM, mod_emb + 1 * DIM, &sm.P[1][0][0], lane);

  zero8(acc);
  gemm_glb(xf, row0, DIM, 4, wgf + 32 * 32, acc, &sm, lane);
  addbias(acc, proj_b + 2 * DIM, 0, lane);
  stage_f32(acc, &sm, lane);
  ln16(&sm, pln_g + 2 * DIM, pln_b + 2 * DIM, mod_emb + 2 * DIM, &sm.P[2][0][0], lane);

  // ---------------- per-stream attention + FFN ----------------
  for (int n = 0; n < 3; ++n) {
    const int kv0 = (n == 0) ? 1 : 0;
    const int kv1 = (n == 2) ? 1 : 2;
    const v16h* Wq = wq + n * 32 * 32;
    const v16h* Wk = wk + n * 32 * 32;
    const v16h* Wv = wv + n * 32 * 32;
    const v16h* Wo = wo + n * 32 * 32;
    const v16h* W1 = wf1 + n * 64 * 32;
    const v16h* W2 = wf2 + n * 64 * 32;

    zero8(acc);
    gemm_lds(&sm.P[n][0][0], 128, 4, Wq, 8, 0, acc, lane);
    addbias(acc, in_b + n * 384 + 0, 0, lane);
    store_h(acc, &sm.q[0][0], 128, 0, lane);

    zero8(acc);
    gemm_lds(&sm.P[kv0][0][0], 128, 4, Wk, 8, 0, acc, lane);
    addbias(acc, in_b + n * 384 + 128, 0, lane);
    store_h(acc, &sm.k0[0][0], 128, 0, lane);

    zero8(acc);
    gemm_lds(&sm.P[kv1][0][0], 128, 4, Wk, 8, 0, acc, lane);
    addbias(acc, in_b + n * 384 + 128, 0, lane);
    store_h(acc, &sm.k1[0][0], 128, 0, lane);

    zero8(acc);
    gemm_lds(&sm.P[kv0][0][0], 128, 4, Wv, 8, 0, acc, lane);
    addbias(acc, in_b + n * 384 + 256, 0, lane);
    store_h(acc, &sm.v0[0][0], 128, 0, lane);

    zero8(acc);
    gemm_lds(&sm.P[kv1][0][0], 128, 4, Wv, 8, 0, acc, lane);
    addbias(acc, in_b + n * 384 + 256, 0, lane);
    store_h(acc, &sm.v1[0][0], 128, 0, lane);
    __syncthreads();

    // attention: 4 heads of 32, 2 keys; lane covers (row = L&15, 2 heads).
    {
      const int r = lane & 15;
#pragma unroll
      for (int hh = 0; hh < 2; ++hh) {
        int base = ((lane >> 4) * 2 + hh) * 32;
        float s0 = 0.f, s1 = 0.f;
        for (int e = 0; e < 32; ++e) {
          float qe = (float)sm.q[r][base + e];
          s0 += qe * (float)sm.k0[r][base + e];
          s1 += qe * (float)sm.k1[r][base + e];
        }
        s0 *= 0.17677669529663687f;  // 1/sqrt(32)
        s1 *= 0.17677669529663687f;
        float m  = fmaxf(s0, s1);
        float e0 = expf(s0 - m), e1 = expf(s1 - m);
        float inv = 1.f / (e0 + e1);
        e0 *= inv; e1 *= inv;
        for (int e = 0; e < 32; ++e) {
          float ov = e0 * (float)sm.v0[r][base + e] + e1 * (float)sm.v1[r][base + e];
          sm.q[r][base + e] = (_Float16)ov;  // o overwrites q
        }
      }
    }
    __syncthreads();

    // out-proj + residual(P) -> LN -> x1 (stored back into sm.q)
    zero8(acc);
    gemm_lds(&sm.q[0][0], 128, 4, Wo, 8, 0, acc, lane);
    addbias(acc, out_b + n * DIM, 0, lane);
    addres_h(acc, &sm.P[n][0][0], lane);
    stage_f32(acc, &sm, lane);
    ln16(&sm, aln_g + n * DIM, aln_b + n * DIM, nullptr, &sm.q[0][0], lane);

    // FFN1 (128 -> 256) in two 8-tile passes; hidden h = [k0|k1] as 16x256.
    _Float16* hbuf = &sm.k0[0][0];
    zero8(acc);
    gemm_lds(&sm.q[0][0], 128, 4, W1, 16, 0, acc, lane);
    addbias(acc, ffn_b1 + n * FFN_DIM, 0, lane);
    gelu8(acc);
    store_h(acc, hbuf, 256, 0, lane);
    zero8(acc);
    gemm_lds(&sm.q[0][0], 128, 4, W1, 16, 8, acc, lane);
    addbias(acc, ffn_b1 + n * FFN_DIM, 128, lane);
    gelu8(acc);
    store_h(acc, hbuf, 256, 128, lane);
    __syncthreads();

    // FFN2 (256 -> 128) + residual(x1) -> LN -> x2[n]
    zero8(acc);
    gemm_lds(hbuf, 256, 8, W2, 8, 0, acc, lane);
    addbias(acc, ffn_b2 + n * DIM, 0, lane);
    addres_h(acc, &sm.q[0][0], lane);
    stage_f32(acc, &sm, lane);
    ln16(&sm, fln_g + n * DIM, fln_b + n * DIM, nullptr, &sm.x2[n][0][0], lane);
  }

  // ---------------- gating + fused output ----------------
  {
    const int r = lane & 15, hp = lane >> 4, c0 = hp * 64;
    float d0 = 0.f, d1 = 0.f, d2 = 0.f;
    for (int n2 = 0; n2 < 3; ++n2) {
      const _Float16* x2r = &sm.x2[n2][r][c0];
      const float* g0 = gate_w + 0 * 384 + n2 * 128 + c0;
      const float* g1 = gate_w + 1 * 384 + n2 * 128 + c0;
      const float* g2 = gate_w + 2 * 384 + n2 * 128 + c0;
      for (int c = 0; c < 64; ++c) {
        float xv = (float)x2r[c];
        d0 += g0[c] * xv; d1 += g1[c] * xv; d2 += g2[c] * xv;
      }
    }
    sm.red[lane][0] = d0; sm.red[lane][1] = d1; sm.red[lane][2] = d2;
    __syncthreads();
    if (lane < 16) {
      float a0 = sm.red[lane][0] + sm.red[lane + 16][0] + gate_b[0];
      float a1 = sm.red[lane][1] + sm.red[lane + 16][1] + gate_b[1];
      float a2 = sm.red[lane][2] + sm.red[lane + 16][2] + gate_b[2];
      float m  = fmaxf(a0, fmaxf(a1, a2));
      float e0 = expf(a0 - m), e1 = expf(a1 - m), e2 = expf(a2 - m);
      float inv = 1.f / (e0 + e1 + e2);
      sm.misc[lane][0] = e0 * inv;
      sm.misc[lane][1] = e1 * inv;
      sm.misc[lane][2] = e2 * inv;
    }
    __syncthreads();
    float g0 = sm.misc[r][0], g1 = sm.misc[r][1], g2 = sm.misc[r][2];
    float* orow = out + (row0 + r) * DIM + c0;
    for (int c = 0; c < 64; c += 4) {
      float4 o4;
      o4.x = g0 * (float)sm.x2[0][r][c0 + c + 0] + g1 * (float)sm.x2[1][r][c0 + c + 0] + g2 * (float)sm.x2[2][r][c0 + c + 0];
      o4.y = g0 * (float)sm.x2[0][r][c0 + c + 1] + g1 * (float)sm.x2[1][r][c0 + c + 1] + g2 * (float)sm.x2[2][r][c0 + c + 1];
      o4.z = g0 * (float)sm.x2[0][r][c0 + c + 2] + g1 * (float)sm.x2[1][r][c0 + c + 2] + g2 * (float)sm.x2[2][r][c0 + c + 2];
      o4.w = g0 * (float)sm.x2[0][r][c0 + c + 3] + g1 * (float)sm.x2[1][r][c0 + c + 3] + g2 * (float)sm.x2[2][r][c0 + c + 3];
      *(float4*)(orow + c) = o4;
    }
  }
}

extern "C" void kernel_launch(void* const* d_in, const int* in_sizes, int n_in,
                              void* d_out, int out_size, void* d_ws, size_t ws_size,
                              hipStream_t stream) {
  const float* xs   = (const float*)d_in[0];
  const float* xg   = (const float*)d_in[1];
  const float* xf   = (const float*)d_in[2];
  const float* pws  = (const float*)d_in[3];   // (128,1280)
  const float* pwgf = (const float*)d_in[4];   // (2,128,128)
  const float* pb   = (const float*)d_in[5];
  const float* plg  = (const float*)d_in[6];
  const float* plb  = (const float*)d_in[7];
  const float* memb = (const float*)d_in[8];
  const float* ipw  = (const float*)d_in[9];   // (3,384,128)
  const float* ipb  = (const float*)d_in[10];
  const float* opw  = (const float*)d_in[11];  // (3,128,128)
  const float* opb  = (const float*)d_in[12];
  const float* alg  = (const float*)d_in[13];
  const float* alb  = (const float*)d_in[14];
  const float* fw1  = (const float*)d_in[15];  // (3,256,128)
  const float* fb1  = (const float*)d_in[16];
  const float* fw2  = (const float*)d_in[17];  // (3,128,256)
  const float* fb2  = (const float*)d_in[18];
  const float* flg  = (const float*)d_in[19];
  const float* flb  = (const float*)d_in[20];
  const float* gw   = (const float*)d_in[21];
  const float* gb   = (const float*)d_in[22];

  const int B = in_sizes[0] / SPATIAL_;

  // ws layout (v16h units; 1 fragment = 32 x v16h = 1 KB). Total 1152 KB.
  v16h* ws  = (v16h*)d_ws;
  v16h* wsp = ws;                 // 320 frags (KB=40, NJ=8)
  v16h* wgf = wsp + 320 * 32;     // 2 x 32
  v16h* wq  = wgf + 64  * 32;     // 3 x 32
  v16h* wk  = wq  + 96  * 32;     // 3 x 32
  v16h* wv  = wk  + 96  * 32;     // 3 x 32
  v16h* wo  = wv  + 96  * 32;     // 3 x 32
  v16h* w1  = wo  + 96  * 32;     // 3 x 64 (KB=4, NJ=16)
  v16h* w2  = w1  + 192 * 32;     // 3 x 64 (KB=8, NJ=8)

  auto swz = [&](const float* src, v16h* dst, int K, int NJ, int KB) {
    int total = KB * NJ * 32;
    swizzle_w<<<(total + 255) / 256, 256, 0, stream>>>(src, dst, K, NJ, KB);
  };

  swz(pws, wsp, SPATIAL_, 8, 40);
  swz(pwgf,            wgf,            DIM, 8, 4);
  swz(pwgf + 128 * 128, wgf + 32 * 32, DIM, 8, 4);
  for (int n = 0; n < 3; ++n) {
    const float* base = ipw + (long)n * 384 * 128;
    swz(base,             wq + n * 32 * 32, DIM, 8, 4);
    swz(base + 128 * 128, wk + n * 32 * 32, DIM, 8, 4);
    swz(base + 256 * 128, wv + n * 32 * 32, DIM, 8, 4);
    swz(opw + (long)n * 128 * 128, wo + n * 32 * 32, DIM, 8, 4);
    swz(fw1 + (long)n * 256 * 128, w1 + n * 64 * 32, DIM, 16, 4);
    swz(fw2 + (long)n * 128 * 256, w2 + n * 64 * 32, FFN_DIM, 8, 8);
  }

  cmaf_fused<<<B / 16, 32, 0, stream>>>(
      xs, xg, xf, pb, plg, plb, memb, ipb, opb, alg, alb,
      fb1, fb2, flg, flb, gw, gb,
      wsp, wgf, wq, wk, wv, wo, w1, w2,
      (float*)d_out);
}
